// ProbAttention_47210280518279
// MI455X (gfx1250) — matrix operand order repaired and checked by reading, not compiled, not run on hardware
//
#include <hip/hip_runtime.h>
#include <math.h>

typedef __attribute__((ext_vector_type(16))) _Float16 v16h;
typedef __attribute__((ext_vector_type(8)))  float    v8f;

#define B_    16
#define H_    8
#define LQ_   4096
#define LK_   4096
#define D_    64
#define NSAMP 45
#define NTOP  45
#define NTOPP 48          // padded rows (3 x 16)
#define PARTS 8
#define CHUNK (LK_ / PARTS)

#define KSTRIDE 72        // halves per K-tile row (144 B, 16-B aligned)
#define VSTRIDE 40        // halves per Vt row    (80 B, 16-B aligned)

// ---- DPP16 xor-lane reductions (pure VALU, no DS traffic) ----
// xor1 = quad_perm[1,0,3,2]=0xB1, xor2 = quad_perm[2,3,0,1]=0x4E,
// xor4 = row_half_mirror=0x141,   xor8 = row_mirror=0x140
template <int CTRL>
__device__ __forceinline__ float dpp_bcast(float v) {
  union { float f; int i; } in, out;
  in.f = v;
  out.i = __builtin_amdgcn_update_dpp(0, in.i, CTRL, 0xf, 0xf, true);
  return out.f;
}
__device__ __forceinline__ float hmax16(float v) {
  v = fmaxf(v, dpp_bcast<0xB1>(v));
  v = fmaxf(v, dpp_bcast<0x4E>(v));
  v = fmaxf(v, dpp_bcast<0x141>(v));
  v = fmaxf(v, dpp_bcast<0x140>(v));
  return v;
}
__device__ __forceinline__ float hsum16(float v) {
  v += dpp_bcast<0xB1>(v);
  v += dpp_bcast<0x4E>(v);
  v += dpp_bcast<0x141>(v);
  v += dpp_bcast<0x140>(v);
  return v;
}

// pack two f32 into two f16 in one dword (v_cvt_pk_f16_f32)
__device__ __forceinline__ unsigned pk2(float a, float b) {
  union { _Float16 h[2]; unsigned u; } t;
  t.h[0] = (_Float16)a; t.h[1] = (_Float16)b;
  return t.u;
}

// load a 16-half B/A fragment slice from 16-byte-aligned LDS (2 x ds_load_b128)
__device__ __forceinline__ v16h ld_frag16(const _Float16* p) {
  union { v16h h; uint4 u[2]; } t;
  t.u[0] = *reinterpret_cast<const uint4*>(p);
  t.u[1] = *reinterpret_cast<const uint4*>(p + 8);
  return t.h;
}

// ============ Kernel A: per-(b,h) mean of V over keys ============
__global__ void vmean_kernel(const float* __restrict__ V, float* __restrict__ vmean) {
  int bh = blockIdx.x, b = bh / H_, h = bh % H_;
  int d = threadIdx.x & 63, kr = threadIdx.x >> 6;  // 4 key-groups x 64 d
  const float* base = V + (((size_t)b * LK_) * H_ + h) * D_ + d;
  float acc = 0.f;
  for (int k = kr; k < LK_; k += 4) acc += base[(size_t)k * H_ * D_];
  __shared__ float red[256];
  red[threadIdx.x] = acc;
  __syncthreads();
  if (kr == 0)
    vmean[bh * D_ + d] = (red[d] + red[64 + d] + red[128 + d] + red[192 + d]) * (1.0f / LK_);
}

// ============ Kernel B: sparsity metric M = max_s(q.k_s) - sum_s(q.k_s)/L_K ============
__global__ void sparm_kernel(const float* __restrict__ Q, const float* __restrict__ K,
                             const int* __restrict__ idx, float* __restrict__ M) {
  int g = threadIdx.x >> 6;         // 4 queries per block, 64 threads each
  int t = threadIdx.x & 63;
  size_t lin = (size_t)blockIdx.x * 4 + g;      // linear over B*H*LQ
  int b = (int)(lin / (H_ * LQ_));
  int rem = (int)(lin % (H_ * LQ_));
  int h = rem / LQ_, q = rem % LQ_;

  __shared__ __align__(16) float qv[4][D_];
  __shared__ float rmax_[4][64];
  __shared__ float rsum_[4][64];

  qv[g][t] = Q[(((size_t)b * LQ_ + q) * H_ + h) * D_ + t];
  __syncthreads();

  float dmax = -INFINITY, dsum = 0.f;
  if (t < NSAMP) {
    int ks = idx[q * NSAMP + t];
    const float4* kp = reinterpret_cast<const float4*>(K + (((size_t)b * LK_ + ks) * H_ + h) * D_);
    const float4* qp = reinterpret_cast<const float4*>(&qv[g][0]);
    float acc = 0.f;
#pragma unroll
    for (int i = 0; i < 16; i++) {
      float4 kv = kp[i], qq = qp[i];
      acc += kv.x * qq.x + kv.y * qq.y + kv.z * qq.z + kv.w * qq.w;
    }
    dmax = acc; dsum = acc;
  }
  rmax_[g][t] = dmax; rsum_[g][t] = dsum;
  __syncthreads();
  for (int off = 32; off > 0; off >>= 1) {
    if (t < off) {
      rmax_[g][t] = fmaxf(rmax_[g][t], rmax_[g][t + off]);
      rsum_[g][t] += rsum_[g][t + off];
    }
    __syncthreads();
  }
  if (t == 0) M[lin] = rmax_[g][0] - rsum_[g][0] * (1.0f / LK_);
}

// ============ Kernel C: top-45 queries per (b,h) (iterative argmax in LDS) ============
__global__ void topk_kernel(const float* __restrict__ M, int* __restrict__ Mtop) {
  int bh = blockIdx.x;
  __shared__ float vals[LQ_];
  __shared__ float rv[256];
  __shared__ int   ri[256];
  for (int i = threadIdx.x; i < LQ_; i += 256) vals[i] = M[(size_t)bh * LQ_ + i];
  __syncthreads();
  for (int it = 0; it < NTOP; ++it) {
    float bv = -INFINITY; int bi = LQ_;
    for (int i = threadIdx.x; i < LQ_; i += 256) {
      float v = vals[i];
      if (v > bv || (v == bv && i < bi)) { bv = v; bi = i; }
    }
    rv[threadIdx.x] = bv; ri[threadIdx.x] = bi;
    __syncthreads();
    for (int off = 128; off > 0; off >>= 1) {
      if (threadIdx.x < off) {
        float ov = rv[threadIdx.x + off]; int oi = ri[threadIdx.x + off];
        if (ov > rv[threadIdx.x] || (ov == rv[threadIdx.x] && oi < ri[threadIdx.x])) {
          rv[threadIdx.x] = ov; ri[threadIdx.x] = oi;
        }
      }
      __syncthreads();
    }
    if (threadIdx.x == 0) { Mtop[bh * NTOP + it] = ri[0]; vals[ri[0]] = -INFINITY; }
    __syncthreads();
  }
}

// ============ Kernel F: broadcast V-mean into the whole output (float4 stream) ============
__global__ void fill_kernel(const float* __restrict__ vmean, float4* __restrict__ out) {
  size_t i = (size_t)blockIdx.x * blockDim.x + threadIdx.x;  // float4 units
  size_t d4 = i & 15;
  size_t h  = (i >> 4) & (H_ - 1);
  size_t bq = i >> 7;           // b*LQ + q
  size_t b  = bq >> 12;         // / LQ_
  const float4* vm = reinterpret_cast<const float4*>(vmean);
  out[i] = vm[(b * H_ + h) * 16 + d4];
}

// ============ Kernel D: flash attention for the 45 top queries, split over 8 K-chunks ====
// 96 threads = 3 waves; wave w owns query rows 16w..16w+15 (rows >= 45 padded with 0).
// Per 32-key step: stage K tile + transposed V tile once per block in LDS (f16),
// then all waves consume fragments via 16-B-aligned ds_load_b128 pairs.
__global__ void attn_kernel(const float* __restrict__ Q, const float* __restrict__ K,
                            const float* __restrict__ V, const int* __restrict__ Mtop,
                            float* __restrict__ Opart, float* __restrict__ mpart,
                            float* __restrict__ lpart) {
  const int w    = threadIdx.x >> 5;
  const int lane = threadIdx.x & 31;
  const int lo   = lane & 15;
  const int hi16 = (lane & 16) ? 1 : 0;
  const int bh   = blockIdx.x / PARTS;
  const int part = blockIdx.x % PARTS;
  const int b = bh / H_, h = bh % H_;
  const int key0 = part * CHUNK;
  const int kofs = hi16 ? 8 : 0;          // unused; A fragments use K offset 16 per lane half

  __shared__ __align__(16) _Float16 Kh[32][KSTRIDE];  // key x d   (f16)
  __shared__ __align__(16) _Float16 Vt[64][VSTRIDE];  // d   x key (f16, transposed)
  __shared__ float ldsP[3][16][33];                   // per-wave P staging

  const float* Kchunk = K + (((size_t)b * LK_ + key0) * H_ + h) * D_;
  const float* Vchunk = V + (((size_t)b * LK_ + key0) * H_ + h) * D_;
  const size_t rowstr = (size_t)H_ * D_;              // floats between key rows

  // ---- Q A-fragments: row = top-query (f32 -> f16) ----
  const int row = w * 16 + lo;
  const int qidx = (row < NTOP) ? Mtop[bh * NTOP + row] : -1;
  v16h a0, a1;
  if (qidx >= 0) {
    const float* qp = Q + (((size_t)b * LQ_ + qidx) * H_ + h) * D_;
#pragma unroll
    for (int j = 0; j < 16; j++) {
      int d = (hi16 ? 16 : 0) + j;          // {0-15} / {16-31}
      a0[j] = (_Float16)qp[d];
      a1[j] = (_Float16)qp[d + 32];
    }
  } else {
#pragma unroll
    for (int j = 0; j < 16; j++) { a0[j] = (_Float16)0.f; a1[j] = (_Float16)0.f; }
  }

  float m_i[8], l_i[8];
#pragma unroll
  for (int r = 0; r < 8; r++) { m_i[r] = -INFINITY; l_i[r] = 0.f; }
  v8f O0 = {}, O1 = {}, O2 = {}, O3 = {};
  const float scale = 0.125f;             // 1/sqrt(64)

  for (int kb = 0; kb < CHUNK; kb += 32) {
    __syncthreads();   // previous iteration done reading Kh/Vt/ldsP

    // ---- cooperative staging: 1024 float4 slots (512 K + 512 V) over 96 threads ----
    for (int slot = threadIdx.x; slot < 1024; slot += 96) {
      int vsel = slot >> 9;               // 0 = K tile, 1 = V tile
      int s    = slot & 511;
      int kr   = s >> 4;                  // key row in tile (0..31)
      int d    = (s & 15) * 4;            // d base (0..60)
      if (vsel == 0) {
        float4 x = *reinterpret_cast<const float4*>(Kchunk + (size_t)(kb + kr) * rowstr + d);
        uint2 pkd; pkd.x = pk2(x.x, x.y); pkd.y = pk2(x.z, x.w);
        *reinterpret_cast<uint2*>(&Kh[kr][d]) = pkd;       // ds_store_b64
      } else {
        float4 x = *reinterpret_cast<const float4*>(Vchunk + (size_t)(kb + kr) * rowstr + d);
        Vt[d + 0][kr] = (_Float16)x.x;                     // transposed store
        Vt[d + 1][kr] = (_Float16)x.y;
        Vt[d + 2][kr] = (_Float16)x.z;
        Vt[d + 3][kr] = (_Float16)x.w;
      }
      // prefetch next chunk into L2 while this one is consumed (global_prefetch_b8)
      if (kb + 32 < CHUNK) {
        const float* nxt = (vsel == 0 ? Kchunk : Vchunk) + (size_t)(kb + 32 + kr) * rowstr + d;
        __builtin_prefetch(nxt, 0, 1);
      }
    }
    __syncthreads();   // staged tiles visible

    // ---- S = Q @ K^T for 32 keys -> two 16x16 f32 tiles (fragments from LDS) ----
    v8f s0, s1;
    {
      v8f z = {};
#pragma unroll
      for (int t = 0; t < 2; t++) {
        const _Float16* kp = &Kh[t * 16 + lo][hi16 ? 16 : 0];
        v16h bk0 = ld_frag16(kp);         // d = (0|16) + j
        v16h bk1 = ld_frag16(kp + 32);    // d = 32 + (0|16) + j
        v8f acc = __builtin_amdgcn_wmma_f32_16x16x32_f16(false, a0, false, bk0,
                                                         (short)0, z, false, false);
        acc     = __builtin_amdgcn_wmma_f32_16x16x32_f16(false, a1, false, bk1,
                                                         (short)0, acc, false, false);
        if (t == 0) s0 = acc; else s1 = acc;
      }
    }

    // ---- online softmax row update (row = r + 8*hi16, across 16 lanes of the half) ----
    float esc[8];
#pragma unroll
    for (int r = 0; r < 8; r++) {
      float x0 = s0[r] * scale, x1 = s1[r] * scale;
      float rm   = hmax16(fmaxf(x0, x1));
      float mnew = fmaxf(m_i[r], rm);
      float p0 = __expf(x0 - mnew);
      float p1 = __expf(x1 - mnew);
      float rs = hsum16(p0 + p1);
      esc[r] = __expf(m_i[r] - mnew);      // exp(-inf)=0 on first block
      l_i[r] = l_i[r] * esc[r] + rs;
      m_i[r] = mnew;
      int Mr = r + hi16 * 8;
      ldsP[w][Mr][lo]      = p0;           // stage P row-major for layout swap
      ldsP[w][Mr][16 + lo] = p1;
    }
#pragma unroll
    for (int r = 0; r < 8; r++) { O0[r] *= esc[r]; O1[r] *= esc[r]; O2[r] *= esc[r]; O3[r] *= esc[r]; }

    __syncthreads();   // P visible across the wave's lanes

    // ---- re-read P as A-fragment (M = lo, K = key-in-chunk) ----
    v16h pa;
#pragma unroll
    for (int j = 0; j < 16; j++) {
      int c = (hi16 ? 16 : 0) + j;
      pa[j] = (_Float16)ldsP[w][lo][c];
    }

    // ---- O += P @ V (4 N-tiles of d, V fragment = contiguous LDS row) ----
#pragma unroll
    for (int t = 0; t < 4; t++) {
      v16h bv = ld_frag16(&Vt[t * 16 + lo][hi16 ? 16 : 0]);  // key = (0|16)+j
      v8f& Ot = (t == 0) ? O0 : (t == 1) ? O1 : (t == 2) ? O2 : O3;
      Ot = __builtin_amdgcn_wmma_f32_16x16x32_f16(false, pa, false, bv,
                                                  (short)0, Ot, false, false);
    }
  }

  // ---- emit split-K partials: un-normalized O, running max m, running sum l ----
  size_t obase = ((size_t)bh * PARTS + part) * NTOPP;
#pragma unroll
  for (int r = 0; r < 8; r++) {
    int rowg = w * 16 + r + hi16 * 8;      // 0..47
    float* op = Opart + (obase + rowg) * D_;
    op[lo]      = O0[r];
    op[16 + lo] = O1[r];
    op[32 + lo] = O2[r];
    op[48 + lo] = O3[r];
    if (lo == 0) { mpart[obase + rowg] = m_i[r]; lpart[obase + rowg] = l_i[r]; }
  }
}

// ============ Kernel E: merge 8 partials (log-sum-exp) and scatter into output ============
__global__ void combine_kernel(const float* __restrict__ Opart, const float* __restrict__ mpart,
                               const float* __restrict__ lpart, const int* __restrict__ Mtop,
                               float* __restrict__ out) {
  int bh = blockIdx.x, b = bh / H_, h = bh % H_;
  int d = threadIdx.x;  // 64
  for (int u = 0; u < NTOP; ++u) {
    float gm = -INFINITY;
#pragma unroll
    for (int p = 0; p < PARTS; p++)
      gm = fmaxf(gm, mpart[((size_t)bh * PARTS + p) * NTOPP + u]);
    float den = 0.f, num = 0.f;
#pragma unroll
    for (int p = 0; p < PARTS; p++) {
      size_t rb = ((size_t)bh * PARTS + p) * NTOPP + u;
      float wgt = __expf(mpart[rb] - gm);
      den += lpart[rb] * wgt;
      num += Opart[rb * D_ + d] * wgt;
    }
    int q = Mtop[bh * NTOP + u];
    out[(((size_t)b * LQ_ + q) * H_ + h) * D_ + d] = num / den;
  }
}

extern "C" void kernel_launch(void* const* d_in, const int* in_sizes, int n_in,
                              void* d_out, int out_size, void* d_ws, size_t ws_size,
                              hipStream_t stream) {
  const float* Q  = (const float*)d_in[0];
  const float* K  = (const float*)d_in[1];
  const float* V  = (const float*)d_in[2];
  const int* idx  = (const int*)d_in[3];
  float* out = (float*)d_out;

  // workspace carve-up (~15.2 MB total)
  float* ws    = (float*)d_ws;
  float* vmean = ws;                                          // B*H*64
  float* Mbuf  = vmean + (size_t)B_ * H_ * D_;                // B*H*LQ
  int*   Mtop  = (int*)(Mbuf + (size_t)B_ * H_ * LQ_);        // B*H*45
  float* Opart = (float*)(Mtop + B_ * H_ * NTOP);             // B*H*8*48*64
  float* mpart = Opart + (size_t)B_ * H_ * PARTS * NTOPP * D_;
  float* lpart = mpart + (size_t)B_ * H_ * PARTS * NTOPP;

  vmean_kernel<<<B_ * H_, 256, 0, stream>>>(V, vmean);
  sparm_kernel<<<(B_ * H_ * LQ_) / 4, 256, 0, stream>>>(Q, K, idx, Mbuf);
  topk_kernel<<<B_ * H_, 256, 0, stream>>>(Mbuf, Mtop);
  fill_kernel<<<(B_ * LQ_ * H_ * D_ / 4) / 256, 256, 0, stream>>>(vmean, (float4*)out);
  attn_kernel<<<B_ * H_ * PARTS, 96, 0, stream>>>(Q, K, V, Mtop, Opart, mpart, lpart);
  combine_kernel<<<B_ * H_, 64, 0, stream>>>(Opart, mpart, lpart, Mtop, out);
}
